// Block_30915174596989
// MI455X (gfx1250) — compile-verified, matrix operand order checked by reading, not run on hardware
//
#include <hip/hip_runtime.h>

#define Bn 2
#define Tn 2048
#define Cn 1024
#define Hn 16
#define HSn 64
#define BT (Bn * Tn)

typedef __bf16 bf16;
typedef __attribute__((ext_vector_type(16))) __bf16 v16bf;
typedef __attribute__((ext_vector_type(8)))  __bf16 v8bf;
typedef __attribute__((ext_vector_type(8)))  float  v8f;

__device__ __forceinline__ bf16 tobf(float f) {
  union { float f; unsigned u; } cv; cv.f = f;
  unsigned short h = (unsigned short)(cv.u >> 16);
  bf16 r; __builtin_memcpy(&r, &h, 2);
  return r;
}

__device__ __forceinline__ v8f vzero8() {
  v8f z;
  #pragma unroll
  for (int i = 0; i < 8; ++i) z[i] = 0.0f;
  return z;
}

// Build a 16-bit A/B fragment per CDNA5 ISA layout:
// lanes 0-15 hold K = {khalf..khalf+7, khalf+16..khalf+23} with khalf = 0,
// lanes 16-31 the same with khalf = 8.
__device__ __forceinline__ v16bf load_frag(const bf16* row, int khalf) {
  v8bf lo = *(const v8bf*)(row + khalf);
  v8bf hi = *(const v8bf*)(row + khalf + 16);
  v16bf r;
  #pragma unroll
  for (int i = 0; i < 8; ++i) { r[i] = lo[i]; r[i + 8] = hi[i]; }
  return r;
}

__device__ __forceinline__ v8f wmma_bf(v16bf a, v16bf b, v8f c) {
  return __builtin_amdgcn_wmma_f32_16x16x32_bf16(false, a, false, b, (short)0, c,
                                                 false, false);
}

// CDNA5 async copy: global -> LDS without VGPR round trip, tracked by ASYNCcnt.
__device__ __forceinline__ void async_ld16(unsigned lds_off,
                                           unsigned long long gaddr) {
  asm volatile("global_load_async_to_lds_b128 %0, %1, off"
               :: "v"(lds_off), "v"(gaddr)
               : "memory");
}
#define WAIT_ASYNCCNT(n) asm volatile("s_wait_asynccnt " n ::: "memory")

// ---------------------------------------------------------------- LayerNorm
__global__ __launch_bounds__(256)
void ln_bf16_kernel(const float* __restrict__ x, const float* __restrict__ w,
                    const float* __restrict__ b, bf16* __restrict__ out) {
  __shared__ float red[256];
  const int tid = threadIdx.x;
  const float* xr = x + (size_t)blockIdx.x * Cn;
  float v[4];
  float s = 0.0f;
  #pragma unroll
  for (int i = 0; i < 4; ++i) { v[i] = xr[tid + i * 256]; s += v[i]; }
  red[tid] = s; __syncthreads();
  for (int o = 128; o > 0; o >>= 1) {
    if (tid < o) red[tid] += red[tid + o];
    __syncthreads();
  }
  const float mu = red[0] * (1.0f / Cn);
  __syncthreads();
  float ss = 0.0f;
  #pragma unroll
  for (int i = 0; i < 4; ++i) { float d = v[i] - mu; ss += d * d; }
  red[tid] = ss; __syncthreads();
  for (int o = 128; o > 0; o >>= 1) {
    if (tid < o) red[tid] += red[tid + o];
    __syncthreads();
  }
  const float inv = rsqrtf(red[0] * (1.0f / Cn) + 1e-5f);
  bf16* orow = out + (size_t)blockIdx.x * Cn;
  #pragma unroll
  for (int i = 0; i < 4; ++i) {
    const int c = tid + i * 256;
    orow[c] = tobf((v[i] - mu) * inv * w[c] + b[c]);
  }
}

// ----------------------------------------------------- weight pack kernels
__global__ void pack_headed_kernel(const float* __restrict__ W,
                                   bf16* __restrict__ Wt) {
  const int idx = blockIdx.x * 256 + threadIdx.x;      // over Cn*Cn
  const int n = idx >> 10;
  const int k = idx & 1023;
  const int hh = n >> 6, d = n & 63;
  Wt[idx] = tobf(W[(((size_t)hh * Cn) + k) * HSn + d]);
}

__global__ void pack_transpose_kernel(const float* __restrict__ W,
                                      bf16* __restrict__ Wt, int K, int N) {
  const int idx = blockIdx.x * 256 + threadIdx.x;
  if (idx >= K * N) return;
  const int n = idx / K;
  const int k = idx - n * K;
  Wt[idx] = tobf(W[(size_t)k * N + n]);
}

// -------------------------------------------------------------- tiled GEMM
// C[M,N] = A[M,K] * Bt[N,K]^T (+bias)(+relu)(+resid). 128x128 block tile,
// 8 waves (2x4), each wave 64x32 = 4x2 WMMA tiles. LDS K-depth 64 -> 16
// WMMAs per barrier pair. Double-buffered async global->LDS copies
// (ASYNCcnt) hide L2 latency; last k-iteration peeled so the steady-state
// loop is branch-free. Dynamic LDS (64 KB): A0@0 B0@16K A1@32K B1@48K.
__device__ __forceinline__ void issue_tile64(const bf16* gA, const bf16* gB,
                                             unsigned ldsA, unsigned ldsB) {
  const unsigned long long a0 = (unsigned long long)(size_t)gA;
  const unsigned long long b0 = (unsigned long long)(size_t)gB;
  #pragma unroll
  for (int i = 0; i < 4; ++i) {
    async_ld16(ldsA + 16u * i, a0 + 16u * i);
    async_ld16(ldsB + 16u * i, b0 + 16u * i);
  }
}

__device__ __forceinline__ void gemm_compute_tile(const bf16* Ab, const bf16* Bb,
                                                  int waveM, int waveN,
                                                  int ln16, int khalf,
                                                  v8f acc[4][2]) {
  #pragma unroll
  for (int ksub = 0; ksub < 64; ksub += 32) {
    v16bf af[4], bfr[2];
    #pragma unroll
    for (int i = 0; i < 4; ++i)
      af[i] = load_frag(Ab + (waveM * 64 + i * 16 + ln16) * 64 + ksub, khalf);
    #pragma unroll
    for (int j = 0; j < 2; ++j)
      bfr[j] = load_frag(Bb + (waveN * 32 + j * 16 + ln16) * 64 + ksub, khalf);
    #pragma unroll
    for (int i = 0; i < 4; ++i)
      #pragma unroll
      for (int j = 0; j < 2; ++j)
        acc[i][j] = wmma_bf(af[i], bfr[j], acc[i][j]);
  }
}

__global__ __launch_bounds__(256)
void gemm_bf16_kernel(const bf16* __restrict__ A, const bf16* __restrict__ Bt,
                      const float* __restrict__ bias,
                      const float* __restrict__ resid,
                      float* __restrict__ outF, bf16* __restrict__ outB,
                      int N, int K, int relu) {
  extern __shared__ char smem[];
  const int tid = threadIdx.x;
  const int lane = tid & 31;
  const int wave = tid >> 5;
  const int waveM = wave >> 2;                 // 0..1
  const int waveN = wave & 3;                  // 0..3
  const int tileM = blockIdx.y * 128;
  const int tileN = blockIdx.x * 128;
  const int ln16 = lane & 15;
  const int khalf = (lane >> 4) * 8;
  const int rowA = tid >> 1;                   // 0..127
  const int colA = (tid & 1) * 32;             // 0 or 32 (elements)

  v8f acc[4][2];
  #pragma unroll
  for (int i = 0; i < 4; ++i)
    #pragma unroll
    for (int j = 0; j < 2; ++j) acc[i][j] = vzero8();

  const unsigned myoff = (unsigned)(rowA * 128 + colA * 2);
  const bf16* gA = A + (size_t)(tileM + rowA) * K + colA;
  const bf16* gB = Bt + (size_t)(tileN + rowA) * K + colA;

  unsigned curA = 0u + myoff, curB = 16384u + myoff;
  unsigned nxtA = 32768u + myoff, nxtB = 49152u + myoff;

  issue_tile64(gA, gB, curA, curB);

  for (int k0 = 0; k0 + 64 < K; k0 += 64) {
    issue_tile64(gA + k0 + 64, gB + k0 + 64, nxtA, nxtB);
    WAIT_ASYNCCNT("0x8");
    __syncthreads();
    gemm_compute_tile((const bf16*)(smem + (curA - myoff)),
                      (const bf16*)(smem + (curB - myoff)),
                      waveM, waveN, ln16, khalf, acc);
    __syncthreads();
    unsigned t;
    t = curA; curA = nxtA; nxtA = t;
    t = curB; curB = nxtB; nxtB = t;
  }
  WAIT_ASYNCCNT("0x0");
  __syncthreads();
  gemm_compute_tile((const bf16*)(smem + (curA - myoff)),
                    (const bf16*)(smem + (curB - myoff)),
                    waveM, waveN, ln16, khalf, acc);

  const int rowOff = (lane < 16) ? 0 : 8;      // C/D layout: lanes>=16 hold M=8..15
  #pragma unroll
  for (int i = 0; i < 4; ++i) {
    #pragma unroll
    for (int j = 0; j < 2; ++j) {
      const int n = tileN + waveN * 32 + j * 16 + ln16;
      const float bv = bias ? bias[n] : 0.0f;
      #pragma unroll
      for (int v = 0; v < 8; ++v) {
        const int m = tileM + waveM * 64 + i * 16 + rowOff + v;
        float val = acc[i][j][v] + bv;
        if (relu) val = fmaxf(val, 0.0f);
        if (resid) val += resid[(size_t)m * N + n];
        if (outF) outF[(size_t)m * N + n] = val;
        else      outB[(size_t)m * N + n] = tobf(val);
      }
    }
  }
}

// --------------------------------------------------------- flash attention
// One block per (b,h, 64 query rows); 4 waves, each wave owns 16 q rows.
// No 1/sqrt(d) scaling (faithful to the reference).
__global__ __launch_bounds__(128)
void attn_kernel(const bf16* __restrict__ Qg, const bf16* __restrict__ Kg,
                 const bf16* __restrict__ Vg, bf16* __restrict__ Og) {
  __shared__ bf16 Klds[64][64];        // [key][c]   == Bt layout for Q*K^T
  __shared__ bf16 Vtlds[64][64];       // [c_out][key] == Bt layout for P*V
  __shared__ bf16 Plds[4][16][64];     // per-wave P staging (C->A relayout)
  const int tid = threadIdx.x;
  const int lane = tid & 31;
  const int w = tid >> 5;
  const int b = blockIdx.x >> 4;
  const int h = blockIdx.x & 15;
  const int qTile = blockIdx.y;
  const int qBase = qTile * 64;
  const int ln16 = lane & 15;
  const int khalf = (lane >> 4) * 8;
  const int rowOff = (lane < 16) ? 0 : 8;

  v16bf qf[2];
  {
    const int t = qBase + w * 16 + ln16;
    const bf16* qrow = Qg + ((size_t)(b * Tn + t)) * Cn + h * HSn;
    qf[0] = load_frag(qrow, khalf);
    qf[1] = load_frag(qrow + 32, khalf);
  }

  v8f o[4];
  #pragma unroll
  for (int i = 0; i < 4; ++i) o[i] = vzero8();
  float mrun[8], lrun[8];
  #pragma unroll
  for (int v = 0; v < 8; ++v) { mrun[v] = -1e30f; lrun[v] = 0.0f; }

  const int ldr = tid >> 1;            // 0..63
  const int ldc = (tid & 1) * 32;      // 0 or 32

  for (int kb = 0; kb <= qTile; ++kb) {
    __syncthreads();
    {
      const bf16* ksrc = Kg + ((size_t)(b * Tn + kb * 64 + ldr)) * Cn + h * HSn + ldc;
      const v8bf* kv = (const v8bf*)ksrc;
      v8bf* kd = (v8bf*)&Klds[ldr][ldc];
      kd[0] = kv[0]; kd[1] = kv[1]; kd[2] = kv[2]; kd[3] = kv[3];
      const bf16* vsrc = Vg + ((size_t)(b * Tn + kb * 64 + ldr)) * Cn + h * HSn + ldc;
      #pragma unroll
      for (int d = 0; d < 32; ++d) Vtlds[ldc + d][ldr] = vsrc[d];
    }
    __syncthreads();

    // S = Q * K^T : 4 column tiles of 16 keys, K=64 in 2 steps.
    v8f s[4];
    #pragma unroll
    for (int ct = 0; ct < 4; ++ct) s[ct] = vzero8();
    #pragma unroll
    for (int ks = 0; ks < 2; ++ks) {
      #pragma unroll
      for (int ct = 0; ct < 4; ++ct) {
        v16bf kf = load_frag(&Klds[ct * 16 + ln16][ks * 32], khalf);
        s[ct] = wmma_bf(qf[ks], kf, s[ct]);
      }
    }

    if (kb == qTile) {                  // causal mask on the diagonal block
      #pragma unroll
      for (int ct = 0; ct < 4; ++ct)
        #pragma unroll
        for (int v = 0; v < 8; ++v)
          if (ct * 16 + ln16 > w * 16 + rowOff + v) s[ct][v] = -1e30f;
    }

    // Online softmax: each lane owns 8 rows; row spread over 16 lanes (cols).
    float pr[4][8];
    #pragma unroll
    for (int v = 0; v < 8; ++v) {
      float mx = fmaxf(fmaxf(s[0][v], s[1][v]), fmaxf(s[2][v], s[3][v]));
      #pragma unroll
      for (int off = 1; off < 16; off <<= 1)
        mx = fmaxf(mx, __shfl_xor(mx, off, 32));
      const float mnew = fmaxf(mrun[v], mx);
      const float scale = __expf(mrun[v] - mnew);
      float rs = 0.0f;
      #pragma unroll
      for (int ct = 0; ct < 4; ++ct) {
        const float p = __expf(s[ct][v] - mnew);
        pr[ct][v] = p;
        rs += p;
      }
      #pragma unroll
      for (int off = 1; off < 16; off <<= 1)
        rs += __shfl_xor(rs, off, 32);
      lrun[v] = lrun[v] * scale + rs;
      mrun[v] = mnew;
      #pragma unroll
      for (int ot = 0; ot < 4; ++ot) o[ot][v] = o[ot][v] * scale;
    }

    // Re-layout P (C-layout) into A-fragment layout through LDS.
    #pragma unroll
    for (int ct = 0; ct < 4; ++ct)
      #pragma unroll
      for (int v = 0; v < 8; ++v)
        Plds[w][rowOff + v][ct * 16 + ln16] = tobf(pr[ct][v]);
    __syncthreads();

    // O += P * V
    #pragma unroll
    for (int ks2 = 0; ks2 < 2; ++ks2) {
      v16bf pf = load_frag(&Plds[w][ln16][ks2 * 32], khalf);
      #pragma unroll
      for (int ot = 0; ot < 4; ++ot) {
        v16bf vf = load_frag(&Vtlds[ot * 16 + ln16][ks2 * 32], khalf);
        o[ot] = wmma_bf(pf, vf, o[ot]);
      }
    }
  }

  #pragma unroll
  for (int ot = 0; ot < 4; ++ot) {
    #pragma unroll
    for (int v = 0; v < 8; ++v) {
      const int t = qBase + w * 16 + rowOff + v;
      Og[((size_t)(b * Tn + t)) * Cn + h * HSn + ot * 16 + ln16] =
          tobf(o[ot][v] / lrun[v]);
    }
  }
}

// ------------------------------------------------------------------- host
extern "C" void kernel_launch(void* const* d_in, const int* in_sizes, int n_in,
                              void* d_out, int out_size, void* d_ws, size_t ws_size,
                              hipStream_t stream) {
  (void)in_sizes; (void)n_in; (void)out_size; (void)ws_size;
  const float* x      = (const float*)d_in[0];
  const float* ln1_w  = (const float*)d_in[1];
  const float* ln1_b  = (const float*)d_in[2];
  const float* Wq     = (const float*)d_in[3];
  const float* Wk     = (const float*)d_in[4];
  const float* Wv     = (const float*)d_in[5];
  const float* proj_w = (const float*)d_in[6];
  const float* proj_b = (const float*)d_in[7];
  const float* ln2_w  = (const float*)d_in[8];
  const float* ln2_b  = (const float*)d_in[9];
  const float* fc1_w  = (const float*)d_in[10];
  const float* fc1_b  = (const float*)d_in[11];
  const float* fc2_w  = (const float*)d_in[12];
  const float* fc2_b  = (const float*)d_in[13];

  char* wsp = (char*)d_ws;
  size_t off = 0;
  auto take = [&](size_t bytes) -> char* {
    char* p = wsp + off;
    off += (bytes + 255) & ~(size_t)255;
    return p;
  };

  bf16* WqT = (bf16*)take((size_t)Cn * Cn * 2);
  bf16* WkT = (bf16*)take((size_t)Cn * Cn * 2);
  bf16* WvT = (bf16*)take((size_t)Cn * Cn * 2);
  bf16* WpT = (bf16*)take((size_t)Cn * Cn * 2);
  bf16* W1T = (bf16*)take((size_t)Cn * 4 * Cn * 2);
  bf16* W2T = (bf16*)take((size_t)Cn * 4 * Cn * 2);
  bf16* Hb  = (bf16*)take((size_t)BT * Cn * 2);       // ln1 out, reused for ln2
  bf16* Qb  = (bf16*)take((size_t)BT * Cn * 2);
  bf16* Kb  = (bf16*)take((size_t)BT * Cn * 2);
  bf16* Vb  = (bf16*)take((size_t)BT * Cn * 2);
  bf16* Ab  = (bf16*)take((size_t)BT * Cn * 2);       // attention output
  float* X1 = (float*)take((size_t)BT * Cn * 4);      // x + sa
  bf16* Fb  = (bf16*)take((size_t)BT * 4 * Cn * 2);   // relu(fc1) out
  float* out = (float*)d_out;

  const size_t GEMM_LDS = 65536;

  // Pack weights to bf16 [N][K].
  pack_headed_kernel<<<Cn * Cn / 256, 256, 0, stream>>>(Wq, WqT);
  pack_headed_kernel<<<Cn * Cn / 256, 256, 0, stream>>>(Wk, WkT);
  pack_headed_kernel<<<Cn * Cn / 256, 256, 0, stream>>>(Wv, WvT);
  pack_transpose_kernel<<<Cn * Cn / 256, 256, 0, stream>>>(proj_w, WpT, Cn, Cn);
  pack_transpose_kernel<<<Cn * 4 * Cn / 256, 256, 0, stream>>>(fc1_w, W1T, Cn, 4 * Cn);
  pack_transpose_kernel<<<Cn * 4 * Cn / 256, 256, 0, stream>>>(fc2_w, W2T, 4 * Cn, Cn);

  // h = LN1(x)
  ln_bf16_kernel<<<BT, 256, 0, stream>>>(x, ln1_w, ln1_b, Hb);

  // Q,K,V = h @ W{q,k,v}
  dim3 gC(Cn / 128, BT / 128);
  gemm_bf16_kernel<<<gC, 256, GEMM_LDS, stream>>>(Hb, WqT, nullptr, nullptr, nullptr, Qb, Cn, Cn, 0);
  gemm_bf16_kernel<<<gC, 256, GEMM_LDS, stream>>>(Hb, WkT, nullptr, nullptr, nullptr, Kb, Cn, Cn, 0);
  gemm_bf16_kernel<<<gC, 256, GEMM_LDS, stream>>>(Hb, WvT, nullptr, nullptr, nullptr, Vb, Cn, Cn, 0);

  // Causal flash attention
  attn_kernel<<<dim3(Bn * Hn, Tn / 64), 128, 0, stream>>>(Qb, Kb, Vb, Ab);

  // x1 = x + attn @ proj_w + proj_b
  gemm_bf16_kernel<<<gC, 256, GEMM_LDS, stream>>>(Ab, WpT, proj_b, x, X1, nullptr, Cn, Cn, 0);

  // h2 = LN2(x1)
  ln_bf16_kernel<<<BT, 256, 0, stream>>>(X1, ln2_w, ln2_b, Hb);

  // ff = relu(h2 @ fc1 + b1); out = x1 + ff @ fc2 + b2
  gemm_bf16_kernel<<<dim3(4 * Cn / 128, BT / 128), 256, GEMM_LDS, stream>>>(
      Hb, W1T, fc1_b, nullptr, nullptr, Fb, 4 * Cn, Cn, 1);
  gemm_bf16_kernel<<<dim3(Cn / 128, BT / 128), 256, GEMM_LDS, stream>>>(
      Fb, W2T, fc2_b, X1, out, nullptr, Cn, 4 * Cn, 0);
}
